// ThreeSIP_vqcm_61976378082044
// MI455X (gfx1250) — matrix operand order repaired and checked by reading, not compiled, not run on hardware
//
#include <hip/hip_runtime.h>
#include <hip/hip_bf16.h>
#include <math.h>

typedef __attribute__((ext_vector_type(16))) _Float16 v16h;
typedef __attribute__((ext_vector_type(8)))  _Float16 v8h;
typedef __attribute__((ext_vector_type(8)))  float    v8f;

union V16 { v16h v; v8h h[2]; };
union U4  { uint4 u; _Float16 h[8]; };

#define LDSA 40   // padded LDS row stride in f16 (80B = 20 banks, conflict-free)

// ---------------------------------------------------------------------------
// Weight convert + zero-pad + K-panel swizzle:
//   dst[((k/32)*Npad + n)*32 + (k%32)] = (k<K && n<N) ? f16(src[k*N+n]) : 0
// so a WMMA B-fragment (one lane, one half) is 16 contiguous f16 in global.
// grid: x covers Npad*32 elems of one panel, y = panel index (k/32)
// ---------------------------------------------------------------------------
__global__ void k_cvt_swz(const float* __restrict__ src, _Float16* __restrict__ dst,
                          int K, int N, int Npad) {
  int i  = blockIdx.x * blockDim.x + threadIdx.x;   // 0 .. Npad*32-1
  int n  = i >> 5;
  int kk = i & 31;
  int k  = blockIdx.y * 32 + kk;
  float v = (k < K && n < N) ? src[(size_t)k * N + n] : 0.0f;
  dst[(size_t)blockIdx.y * Npad * 32 + i] = (_Float16)v;
}

// ---------------------------------------------------------------------------
// Fused GEMM:  act( A[M x K] * W[K x N] + bias )
//   A: f32 (AF32, chunk-level K guard; K%8==0) or f16 (padded, stride lda)
//   Wp: f16 panel-swizzled (see k_cvt_swz); B frags read direct from global/L2
//   outF (f32, ld ldf) and/or outH (f16, ld ldh), ELU optional.
// 128x128 block tile, BK=32, 256 threads (8 waves), wave tile 32x64,
// double-buffered LDS A tile (padded stride), v_wmma_f32_16x16x32_f16.
// ---------------------------------------------------------------------------
template <bool AF32>
__global__ __launch_bounds__(256) void k_gemm(
    const void* __restrict__ Aptr, int lda, int Ktrue, int Kiters,
    const _Float16* __restrict__ Wp, int Npad,
    const float* __restrict__ bias, int N,
    float* __restrict__ outF, int ldf,
    _Float16* __restrict__ outH, int ldh,
    int doElu)
{
  __shared__ __align__(32) _Float16 lA[2][128 * LDSA];   // double-buffered [m][k]

  const int t    = threadIdx.x;
  const int lane = t & 31;
  const int wave = t >> 5;
  const int wm   = (wave & 3) * 32;   // 4 waves down M
  const int wn   = (wave >> 2) * 64;  // 2 waves across N
  const int hl   = lane >> 4;         // half-wave select
  const int lm   = lane & 15;

  const int m0 = blockIdx.y * 128;
  const int n0 = blockIdx.x * 128;

  const float*    A32 = (const float*)Aptr;
  const _Float16* A16 = (const _Float16*)Aptr;

  // staging indices for this thread (2 chunks of 8 f16)
  const int row0 = (t * 2) >> 2;
  const int c80  = ((t * 2) & 3) * 8;
  const int row1 = (t * 2 + 1) >> 2;
  const int c81  = ((t * 2 + 1) & 3) * 8;

  auto loadA = [&](int kt, U4 reg[2]) {
    const int rows[2] = {row0, row1};
    const int c8s[2]  = {c80, c81};
#pragma unroll
    for (int i = 0; i < 2; ++i) {
      int gk = kt * 32 + c8s[i];
      if (AF32) {
        if (gk + 8 <= Ktrue) {            // K % 8 == 0 -> whole-chunk guard
          const float* s = A32 + (size_t)(m0 + rows[i]) * lda + gk;
          float4 f0 = *(const float4*)(s);
          float4 f1 = *(const float4*)(s + 4);
          reg[i].h[0] = (_Float16)f0.x; reg[i].h[1] = (_Float16)f0.y;
          reg[i].h[2] = (_Float16)f0.z; reg[i].h[3] = (_Float16)f0.w;
          reg[i].h[4] = (_Float16)f1.x; reg[i].h[5] = (_Float16)f1.y;
          reg[i].h[6] = (_Float16)f1.z; reg[i].h[7] = (_Float16)f1.w;
        } else {
          reg[i].u = make_uint4(0u, 0u, 0u, 0u);
        }
      } else {
        reg[i].u = *(const uint4*)&A16[(size_t)(m0 + rows[i]) * lda + gk];
      }
    }
  };
  auto storeA = [&](int buf, U4 reg[2]) {
    *(uint4*)&lA[buf][row0 * LDSA + c80] = reg[0].u;
    *(uint4*)&lA[buf][row1 * LDSA + c81] = reg[1].u;
  };

  v8f zero = {};
  v8f acc[2][4];
#pragma unroll
  for (int i = 0; i < 2; ++i)
#pragma unroll
    for (int j = 0; j < 4; ++j) acc[i][j] = zero;

  U4 reg[2];
  loadA(0, reg);
  storeA(0, reg);

  for (int kt = 0; kt < Kiters; ++kt) {
    __syncthreads();
    const bool more = (kt + 1) < Kiters;
    if (more) loadA(kt + 1, reg);       // prefetch next A tile into regs

    // B fragments straight from global (panel-swizzled, contiguous per lane)
    const _Float16* panel = Wp + (size_t)kt * Npad * 32;
    v16h bf[4];
#pragma unroll
    for (int nt = 0; nt < 4; ++nt)
      bf[nt] = *(const v16h*)&panel[(size_t)(n0 + wn + nt * 16 + lm) * 32 + hl * 16];

    // A fragments from LDS (contiguous k = 16*half + s, matching B panels)
    V16 af[2];
#pragma unroll
    for (int mt = 0; mt < 2; ++mt) {
      int r = wm + mt * 16 + lm;
      af[mt].h[0] = *(const v8h*)&lA[kt & 1][r * LDSA + hl * 16];
      af[mt].h[1] = *(const v8h*)&lA[kt & 1][r * LDSA + hl * 16 + 8];
    }
#pragma unroll
    for (int mt = 0; mt < 2; ++mt)
#pragma unroll
      for (int nt = 0; nt < 4; ++nt)
        acc[mt][nt] = __builtin_amdgcn_wmma_f32_16x16x32_f16(
            false, af[mt].v, false, bf[nt], (short)0, acc[mt][nt], false, false);

    if (more) storeA((kt + 1) & 1, reg);
  }

  // ---- epilogue: bias + ELU, write f32 and/or f16 ----
#pragma unroll
  for (int mt = 0; mt < 2; ++mt) {
#pragma unroll
    for (int nt = 0; nt < 4; ++nt) {
      int col = n0 + wn + nt * 16 + lm;
      if (col >= N) continue;
      float bv = bias[col];
#pragma unroll
      for (int v = 0; v < 8; ++v) {
        int rowg = m0 + wm + mt * 16 + hl * 8 + v;
        float x = acc[mt][nt][v] + bv;
        if (doElu) x = (x > 0.0f) ? x : expm1f(x);
        if (outF) outF[(size_t)rowg * ldf + col] = x;
        if (outH) outH[(size_t)rowg * ldh + col] = (_Float16)x;
      }
    }
  }
}

// ---------------------------------------------------------------------------
// VQ nearest-code: one thread per 16-dim group. argmin_c(|e_c|^2 - 2 z.e_c)
// ---------------------------------------------------------------------------
__global__ void k_vq(const float* __restrict__ ze, const float* __restrict__ emb,
                     float* __restrict__ zq, _Float16* __restrict__ zqh, int total)
{
  __shared__ float se[128 * 16];
  __shared__ float sn[128];
  for (int i = threadIdx.x; i < 128 * 16; i += blockDim.x) se[i] = emb[i];
  __syncthreads();
  if (threadIdx.x < 128) {
    float s = 0.0f;
#pragma unroll
    for (int d = 0; d < 16; ++d) { float e = se[threadIdx.x * 16 + d]; s += e * e; }
    sn[threadIdx.x] = s;
  }
  __syncthreads();

  int id = blockIdx.x * blockDim.x + threadIdx.x;
  if (id >= total) return;
  const float* z = ze + (size_t)id * 16;
  float zr[16];
#pragma unroll
  for (int d = 0; d < 16; ++d) zr[d] = z[d];

  int best = 0;
  float bestd = 3.4e38f;
  for (int c = 0; c < 128; ++c) {
    float dot = 0.0f;
#pragma unroll
    for (int d = 0; d < 16; ++d) dot += zr[d] * se[c * 16 + d];
    float dist = sn[c] - 2.0f * dot;
    if (dist < bestd) { bestd = dist; best = c; }   // first-min like argmin
  }

  float*    o  = zq  + (size_t)id * 16;
  _Float16* oh = zqh + (size_t)id * 16;
#pragma unroll
  for (int d = 0; d < 16; ++d) { float e = se[best * 16 + d]; o[d] = e; oh[d] = (_Float16)e; }
}

// ---------------------------------------------------------------------------
extern "C" void kernel_launch(void* const* d_in, const int* in_sizes, int n_in,
                              void* d_out, int out_size, void* d_ws, size_t ws_size,
                              hipStream_t stream)
{
  (void)in_sizes; (void)n_in; (void)out_size; (void)ws_size;
  const float* y   = (const float*)d_in[0];
  const float* emb = (const float*)d_in[1];
  const float* eW1 = (const float*)d_in[2];
  const float* eb1 = (const float*)d_in[3];
  const float* eW2 = (const float*)d_in[4];
  const float* eb2 = (const float*)d_in[5];
  const float* eW3 = (const float*)d_in[6];
  const float* eb3 = (const float*)d_in[7];
  const float* dW1 = (const float*)d_in[8];
  const float* db1 = (const float*)d_in[9];
  const float* dW2 = (const float*)d_in[10];
  const float* db2 = (const float*)d_in[11];
  const float* dW3 = (const float*)d_in[12];
  const float* db3 = (const float*)d_in[13];

  const int M = 12800;                 // B * n_m
  char* ws = (char*)d_ws;
  size_t off = 0;
  auto carve = [&](size_t bytes) { void* p = ws + off; off += bytes; return p; };
  _Float16* eW1h = (_Float16*)carve((size_t)2528 * 1024 * 2);
  _Float16* eW2h = (_Float16*)carve((size_t)1024 * 1024 * 2);
  _Float16* eW3h = (_Float16*)carve((size_t)1024 * 2048 * 2);
  _Float16* dW1h = (_Float16*)carve((size_t)2048 * 1024 * 2);
  _Float16* dW2h = (_Float16*)carve((size_t)1024 * 1024 * 2);
  _Float16* dW3h = (_Float16*)carve((size_t)1024 * 2560 * 2);
  _Float16* actA = (_Float16*)carve((size_t)M * 1024 * 2);   // z1h, then d1h
  _Float16* actB = (_Float16*)carve((size_t)M * 1024 * 2);   // z2h, then d2h
  _Float16* zqh  = (_Float16*)carve((size_t)M * 2048 * 2);

  // d_out = [out (12800x2520), ze (12800x2048), zq (12800x2048)]
  float* outF = (float*)d_out;
  float* zeF  = outF + (size_t)M * 2520;
  float* zqF  = zeF  + (size_t)M * 2048;

  auto cvt = [&](const float* src, _Float16* dst, int K, int N, int Kpad, int Npad) {
    dim3 g((Npad * 32) / 256, Kpad / 32);
    k_cvt_swz<<<g, 256, 0, stream>>>(src, dst, K, N, Npad);
  };
  cvt(eW1, eW1h, 2520, 1024, 2528, 1024);
  cvt(eW2, eW2h, 1024, 1024, 1024, 1024);
  cvt(eW3, eW3h, 1024, 2048, 1024, 2048);
  cvt(dW1, dW1h, 2048, 1024, 2048, 1024);
  cvt(dW2, dW2h, 1024, 1024, 1024, 1024);
  cvt(dW3, dW3h, 1024, 2520, 1024, 2560);

  dim3 blk(256);
  // encoder
  k_gemm<true ><<<dim3( 8, 100), blk, 0, stream>>>(y,    2520, 2520, 79, eW1h, 1024,
      eb1, 1024, nullptr, 0, actA, 1024, 1);
  k_gemm<false><<<dim3( 8, 100), blk, 0, stream>>>(actA, 1024, 1024, 32, eW2h, 1024,
      eb2, 1024, nullptr, 0, actB, 1024, 1);
  k_gemm<false><<<dim3(16, 100), blk, 0, stream>>>(actB, 1024, 1024, 32, eW3h, 2048,
      eb3, 2048, zeF, 2048, nullptr, 0, 0);
  // vector quantization (straight-through => decoder input is zq)
  k_vq<<<dim3((M * 128) / 256), blk, 0, stream>>>(zeF, emb, zqF, zqh, M * 128);
  // decoder
  k_gemm<false><<<dim3( 8, 100), blk, 0, stream>>>(zqh,  2048, 2048, 64, dW1h, 1024,
      db1, 1024, nullptr, 0, actA, 1024, 1);
  k_gemm<false><<<dim3( 8, 100), blk, 0, stream>>>(actA, 1024, 1024, 32, dW2h, 1024,
      db2, 1024, nullptr, 0, actB, 1024, 1);
  k_gemm<false><<<dim3(20, 100), blk, 0, stream>>>(actB, 1024, 1024, 32, dW3h, 2560,
      db3, 2520, outF, 2520, nullptr, 0, 0);
}
